// DualAttention_55001351192730
// MI455X (gfx1250) — matrix operand order repaired and checked by reading, not compiled, not run on hardware
//
#include <hip/hip_runtime.h>
#include <cstdint>
#include <cstddef>

// ---------------- problem constants ----------------
constexpr int kB  = 2;
constexpr int kS  = 2048;
constexpr int kD  = 1024;   // d_model
constexpr int kH  = 8;      // heads per branch
constexpr int kHD = 64;     // head dim
constexpr int kP  = 512;    // per-branch projection width (kH*kHD)
constexpr int kM  = kB * kS; // 4096 token rows

typedef __bf16 bf16_t;
typedef __attribute__((ext_vector_type(16))) __bf16 v16bf;
typedef __attribute__((ext_vector_type(8)))  __bf16 v8bf;
typedef __attribute__((ext_vector_type(8)))  float  v8f;
typedef uint32_t u32x4 __attribute__((ext_vector_type(4)));
typedef uint32_t u32x8 __attribute__((ext_vector_type(8)));

// ---------------- WMMA fragment helpers (wave32, 16x16x32 bf16) -------------
// A fragment: 16(M) x 32(K).  lane<16: M=lane, K={0..7,16..23}; lane>=16: M=lane-16, K={8..15,24..31}
static __device__ __forceinline__ v16bf load_A_frag(const bf16_t* p, int stride, int lane) {
  const int m  = lane & 15;
  const int c0 = (lane >> 4) << 3;            // 0 or 8
  const bf16_t* r = p + m * stride;
  v8bf lo = *reinterpret_cast<const v8bf*>(r + c0);
  v8bf hi = *reinterpret_cast<const v8bf*>(r + c0 + 16);
  v16bf out;
#pragma unroll
  for (int t = 0; t < 8; ++t) { out[t] = lo[t]; out[t + 8] = hi[t]; }
  return out;
}

// B fragment: 32(K) x 16(N) where memory holds B^T rows, i.e. p[n*stride + k].
// lane<16: N=lane, K={0..7,16..23}; lane>=16: N=lane-16, K={8..15,24..31}
static __device__ __forceinline__ v16bf load_B_frag(const bf16_t* p, int stride, int lane) {
  const int n  = lane & 15;
  const int kb = (lane >> 4) << 3;            // 0 or 8
  const bf16_t* r = p + n * stride + kb;
  v8bf lo = *reinterpret_cast<const v8bf*>(r);
  v8bf hi = *reinterpret_cast<const v8bf*>(r + 16);
  v16bf out;
#pragma unroll
  for (int t = 0; t < 8; ++t) { out[t] = lo[t]; out[t + 8] = hi[t]; }
  return out;
}

static __device__ __forceinline__ v8f wmma_bf16(v16bf a, v16bf b, v8f c) {
  return __builtin_amdgcn_wmma_f32_16x16x32_bf16(false, a, false, b, (short)0, c, false, false);
}

// reductions across the 16 lanes that share one C-tile row (xor 1,2,4,8 stays in half)
static __device__ __forceinline__ float row_max16(float v) {
#pragma unroll
  for (int off = 1; off < 16; off <<= 1) v = fmaxf(v, __shfl_xor(v, off, 32));
  return v;
}
static __device__ __forceinline__ float row_sum16(float v) {
#pragma unroll
  for (int off = 1; off < 16; off <<= 1) v += __shfl_xor(v, off, 32);
  return v;
}

// ---------------- CDNA5 async / TDM data movement ----------------
static __device__ __forceinline__ uint32_t lds_addr32(const void* p) {
  return (uint32_t)(uintptr_t)p;  // LDS aperture: low 32 bits are the DS byte address
}

// async 16B copy global->LDS, tracked by ASYNCcnt
static __device__ __forceinline__ void async_copy_b128(uint32_t lds_byte, const bf16_t* g) {
  asm volatile("global_load_async_to_lds_b128 %0, %1, off"
               :: "v"(lds_byte), "v"(g)
               : "memory");
}
static __device__ __forceinline__ void wait_asynccnt0() {
  asm volatile("s_wait_asynccnt 0x0" ::: "memory");
}
static __device__ __forceinline__ void wait_asynccnt2() {
  asm volatile("s_wait_asynccnt 0x2" ::: "memory");  // allow next tile's 2 copies in flight
}

// Tensor Data Mover: DMA a rows x cols (bf16) 2D tile, row stride = stride_elems,
// into LDS at lds_byte with 4-DWORD padding after every 16 DWORDs
// (=> effective LDS row stride of 40 bf16 elements for a 32-element tile row).
static __device__ __forceinline__ void tdm_load_tile_2d(uint32_t lds_byte, const bf16_t* g,
                                                        uint32_t rows, uint32_t cols,
                                                        uint32_t stride_elems,
                                                        uint32_t tensor_d0, uint32_t tensor_d1) {
  const uint64_t ga = (uint64_t)(uintptr_t)g;
  u32x4 g0;
  g0[0] = 1u;                                              // count=1, user descriptor
  g0[1] = lds_byte;                                        // lds_addr
  g0[2] = (uint32_t)(ga & 0xffffffffu);                    // global_addr[31:0]
  g0[3] = (uint32_t)((ga >> 32) & 0x01ffffffu) | (2u << 30); // global_addr[56:32], type=2
  u32x8 g1;
  g1[0] = (1u << 16)        // data_size = 2 bytes
        | (1u << 20)        // pad_enable
        | (3u << 22)        // pad_interval: 16 DWORDs
        | (3u << 25);       // pad_amount:   4 DWORDs
  g1[1] = (tensor_d0 & 0xffffu) << 16;                                    // tensor_dim0[15:0]
  g1[2] = ((tensor_d0 >> 16) & 0xffffu) | ((tensor_d1 & 0xffffu) << 16);  // dim0 hi | dim1 lo
  g1[3] = ((tensor_d1 >> 16) & 0xffffu) | ((cols & 0xffffu) << 16);       // dim1 hi | tile_dim0
  g1[4] = rows & 0xffffu;                                                 // tile_dim1 (tile_dim2=0)
  g1[5] = stride_elems;                                                   // tensor_dim0_stride lo
  g1[6] = 0u;
  g1[7] = 0u;
  asm volatile("tensor_load_to_lds %0, %1" :: "s"(g0), "s"(g1) : "memory");
}

// ---------------- elementwise kernels ----------------
__global__ __launch_bounds__(256) void f32_to_bf16_kernel(const float* __restrict__ in,
                                                          bf16_t* __restrict__ out, int n) {
  int i = blockIdx.x * 256 + threadIdx.x;
  if (i < n) out[i] = (bf16_t)in[i];
}

// RoPE over [kM, kH, kHD] bf16; freqs [kS, 32] f32. one thread per (row, head, pair)
__global__ __launch_bounds__(256) void rope_kernel(const bf16_t* __restrict__ in,
                                                   const float* __restrict__ cosb,
                                                   const float* __restrict__ sinb,
                                                   bf16_t* __restrict__ out) {
  int gid  = blockIdx.x * 256 + threadIdx.x;  // kM*256 total
  int rowi = gid >> 8;
  int rem  = gid & 255;
  int hh   = rem >> 5;
  int d2   = rem & 31;
  int s    = rowi & (kS - 1);
  size_t base = (size_t)rowi * kP + hh * kHD + d2 * 2;
  float xr = (float)in[base];
  float xi = (float)in[base + 1];
  float c  = cosb[s * 32 + d2];
  float sn = sinb[s * 32 + d2];
  out[base]     = (bf16_t)(xr * c - xi * sn);
  out[base + 1] = (bf16_t)(xr * sn + xi * c);
}

// ---------------- GEMM: C[M,N] = A[M,K] * W[N,K]^T  (bf16 in, f32 acc) -------
// workgroup tile 64x64, 4 waves of 32x32 (2x2 WMMA tiles each), K-step 32.
// Double-buffered LDS staging: A tile via GLOBAL_LOAD_ASYNC_TO_LDS_B128 (ASYNCcnt),
// W tile via the Tensor Data Mover (TENSORcnt) with LDS pad -> stride 40.
// Steady state keeps the next tile's DMA in flight under the current tile's WMMAs
// (s_wait_asynccnt 2 / s_wait_tensorcnt 1), draining to zero only on the last tile.
template<bool OUT_F32>
__global__ __launch_bounds__(128) void gemm_bf16_wmma(const bf16_t* __restrict__ A,
                                                      const bf16_t* __restrict__ W,
                                                      float* __restrict__ Cf,
                                                      bf16_t* __restrict__ Cb,
                                                      int K, int ldc, int col_off) {
  __shared__ bf16_t As[2][64 * 40];
  __shared__ bf16_t Ws[2][64 * 40];

  const int tid  = threadIdx.x;
  const int lane = tid & 31;
  const int wave = tid >> 5;
  const int wm   = (wave >> 1) * 32;
  const int wn   = (wave & 1) * 32;
  const size_t bm = (size_t)blockIdx.x * 64;
  const size_t bn = (size_t)blockIdx.y * 64;

  v8f acc[2][2] = {};

  const int row  = tid >> 2;        // 0..31
  const int col8 = (tid & 3) * 8;   // 0,8,16,24

  const uint32_t ldsA0[2] = { lds_addr32(&As[0][row * 40 + col8]),
                              lds_addr32(&As[1][row * 40 + col8]) };
  const uint32_t ldsA1[2] = { lds_addr32(&As[0][(row + 32) * 40 + col8]),
                              lds_addr32(&As[1][(row + 32) * 40 + col8]) };
  const uint32_t ldsW[2]  = { lds_addr32(Ws[0]), lds_addr32(Ws[1]) };
  const bf16_t* gA0 = A + (bm + row) * K + col8;
  const bf16_t* gA1 = A + (bm + row + 32) * K + col8;
  const bf16_t* gW  = W + bn * K;

  auto stage = [&](int buf, int k0) {
    async_copy_b128(ldsA0[buf], gA0 + k0);
    async_copy_b128(ldsA1[buf], gA1 + k0);
    if (wave == 0) {
      tdm_load_tile_2d(ldsW[buf], gW + k0, 64u, 32u, (uint32_t)K,
                       (uint32_t)K, 1u << 20);
    }
  };

  const int nk = K >> 5;
  stage(0, 0);  // prologue

  for (int i = 0; i < nk; ++i) {
    const int buf = i & 1;
    if (i + 1 < nk) {
      stage(buf ^ 1, (i + 1) << 5);        // prefetch next tile into other buffer
      wait_asynccnt2();                    // current tile's 2 copies complete
      if (wave == 0) __builtin_amdgcn_s_wait_tensorcnt(1);
    } else {
      wait_asynccnt0();                    // tail: drain everything
      if (wave == 0) __builtin_amdgcn_s_wait_tensorcnt(0);
    }
    __syncthreads();
#pragma unroll
    for (int mi = 0; mi < 2; ++mi) {
      v16bf a = load_A_frag(&As[buf][(wm + mi * 16) * 40], 40, lane);
#pragma unroll
      for (int ni = 0; ni < 2; ++ni) {
        v16bf b = load_B_frag(&Ws[buf][(wn + ni * 16) * 40], 40, lane);
        acc[mi][ni] = wmma_bf16(a, b, acc[mi][ni]);
      }
    }
    __syncthreads();  // all reads of buf done before it is re-staged next iteration
  }

  const int nl  = lane & 15;
  const int mhi = (lane >> 4) * 8;
#pragma unroll
  for (int mi = 0; mi < 2; ++mi) {
#pragma unroll
    for (int ni = 0; ni < 2; ++ni) {
      const size_t col = bn + wn + ni * 16 + nl + col_off;
#pragma unroll
      for (int v = 0; v < 8; ++v) {
        const size_t r = bm + wm + mi * 16 + mhi + v;
        const float val = acc[mi][ni][v];
        if (OUT_F32) Cf[r * ldc + col] = val;
        else         Cb[r * ldc + col] = (bf16_t)val;
      }
    }
  }
}

// ---------------- fused causal flash attention (+ optional relational term) --
// grid: (kB*kH, kS/64). 4 waves per WG, each owns a 16-row query tile.
template<bool WITH_REL>
__global__ __launch_bounds__(128) void flash_attn_kernel(const bf16_t* __restrict__ Q,
                                                         const bf16_t* __restrict__ Kk,
                                                         const bf16_t* __restrict__ V,
                                                         const bf16_t* __restrict__ QR,
                                                         const bf16_t* __restrict__ KR,
                                                         const float*  __restrict__ WR,
                                                         bf16_t* __restrict__ Out) {
  __shared__ bf16_t Vt[64 * 40];      // V^T tile: [d][j], j-step = 32
  __shared__ bf16_t Ps[4][16 * 40];   // per-wave P scratch (C->A layout bounce)

  const int bh   = blockIdx.x;
  const int b    = bh >> 3;
  const int h    = bh & 7;
  const int tid  = threadIdx.x;
  const int lane = tid & 31;
  const int wave = tid >> 5;
  const int i0   = blockIdx.y * 64 + wave * 16;

  const float scale = 0.125f;  // 1/sqrt(64)

  const bf16_t* Qbase = Q + ((size_t)(b * kS + i0)) * kP + h * kHD;
  const v16bf aq0 = load_A_frag(Qbase, kP, lane);       // head-dim K 0..31
  const v16bf aq1 = load_A_frag(Qbase + 32, kP, lane);  // head-dim K 32..63

  float mrow[8], lrow[8];
#pragma unroll
  for (int v = 0; v < 8; ++v) { mrow[v] = -3.0e38f; lrow[v] = 0.f; }
  v8f o_acc[4] = {};
  v8f racc[WITH_REL ? 8 : 1] = {};

  const int nl   = lane & 15;
  const int mhi  = (lane >> 4) * 8;
  const int jend = blockIdx.y * 64 + 64;

  for (int j0 = 0; j0 < jend; j0 += 32) {
    // cooperative transpose-stage of V tile (32 rows x 64 dims -> Vt[d][j])
    {
      const int jr = tid >> 2;
      const int c0 = (tid & 3) * 16;
      const bf16_t* vg = V + ((size_t)(b * kS + j0 + jr)) * kP + h * kHD + c0;
      v8bf a0 = *reinterpret_cast<const v8bf*>(vg);
      v8bf a1 = *reinterpret_cast<const v8bf*>(vg + 8);
#pragma unroll
      for (int t = 0; t < 8; ++t) {
        Vt[(c0 + t) * 40 + jr]     = a0[t];
        Vt[(c0 + 8 + t) * 40 + jr] = a1[t];
      }
    }
    __syncthreads();

    if (j0 < i0 + 16) {  // uniform per wave (causal participation)
      // ---- scores: Q(16x64) . K^T -> 16x32 ----
      v8f sc[2] = {};
#pragma unroll
      for (int jt = 0; jt < 2; ++jt) {
        const bf16_t* Kb = Kk + ((size_t)(b * kS + j0 + jt * 16)) * kP + h * kHD;
        sc[jt] = wmma_bf16(aq0, load_B_frag(Kb, kP, lane), sc[jt]);
        sc[jt] = wmma_bf16(aq1, load_B_frag(Kb + 32, kP, lane), sc[jt]);
      }
      // ---- causal mask + scale ----
#pragma unroll
      for (int jt = 0; jt < 2; ++jt) {
        const int jg = j0 + jt * 16 + nl;
#pragma unroll
        for (int v = 0; v < 8; ++v) {
          const int ig = i0 + mhi + v;
          sc[jt][v] = (jg <= ig) ? sc[jt][v] * scale : -3.0e38f;
        }
      }
      // ---- online softmax ----
      float corr[8];
#pragma unroll
      for (int v = 0; v < 8; ++v) {
        const float sm = row_max16(fmaxf(sc[0][v], sc[1][v]));
        const float mn = fmaxf(mrow[v], sm);
        corr[v] = __expf(mrow[v] - mn);
        mrow[v] = mn;
      }
#pragma unroll
      for (int v = 0; v < 8; ++v) {
        const float p0 = __expf(sc[0][v] - mrow[v]);
        const float p1 = __expf(sc[1][v] - mrow[v]);
        sc[0][v] = p0; sc[1][v] = p1;
        lrow[v] = lrow[v] * corr[v] + row_sum16(p0 + p1);
      }
#pragma unroll
      for (int dt = 0; dt < 4; ++dt)
#pragma unroll
        for (int v = 0; v < 8; ++v) o_acc[dt][v] *= corr[v];
      if constexpr (WITH_REL) {
#pragma unroll
        for (int r = 0; r < 8; ++r)
#pragma unroll
          for (int v = 0; v < 8; ++v) racc[r][v] *= corr[v];
      }
      // ---- P (C layout) -> LDS bf16 -> A fragment ----
      bf16_t* Pw = Ps[wave];
#pragma unroll
      for (int jt = 0; jt < 2; ++jt)
#pragma unroll
        for (int v = 0; v < 8; ++v)
          Pw[(mhi + v) * 40 + jt * 16 + nl] = (bf16_t)sc[jt][v];
      const v16bf ap = load_A_frag(Pw, 40, lane);
      // ---- O += P . V ----
#pragma unroll
      for (int dt = 0; dt < 4; ++dt)
        o_acc[dt] = wmma_bf16(ap, load_B_frag(&Vt[dt * 16 * 40], 40, lane), o_acc[dt]);

      // ---- relational contribution ----
      if constexpr (WITH_REL) {
        const float rscale = 0.125f;  // 1/sqrt(REL_PROJ=64)
        const bf16_t* QRb = QR + ((size_t)(b * kS + i0)) * kP;
        const bf16_t* KRb = KR + ((size_t)(b * kS + j0)) * kP;
#pragma unroll
        for (int r = 0; r < 8; ++r) {
          const v16bf ar0 = load_A_frag(QRb + r * 64, kP, lane);
          const v16bf ar1 = load_A_frag(QRb + r * 64 + 32, kP, lane);
#pragma unroll
          for (int jt = 0; jt < 2; ++jt) {
            const bf16_t* kb = KRb + (size_t)jt * 16 * kP + r * 64;
            v8f rel = {};
            rel = wmma_bf16(ar0, load_B_frag(kb, kP, lane), rel);
            rel = wmma_bf16(ar1, load_B_frag(kb + 32, kP, lane), rel);
#pragma unroll
            for (int v = 0; v < 8; ++v)
              racc[r][v] += sc[jt][v] * rel[v] * rscale;
          }
        }
      }
    }
    __syncthreads();
  }

  float inv_l[8];
#pragma unroll
  for (int v = 0; v < 8; ++v) inv_l[v] = 1.0f / lrow[v];

  if constexpr (WITH_REL) {
#pragma unroll
    for (int r = 0; r < 8; ++r)
#pragma unroll
      for (int v = 0; v < 8; ++v) racc[r][v] = row_sum16(racc[r][v]);
  }

  bf16_t* ob = Out + ((size_t)(b * kS + i0)) * kP + h * kHD;
#pragma unroll
  for (int dt = 0; dt < 4; ++dt) {
    const int d = dt * 16 + nl;
    float wr8[8] = {};
    if constexpr (WITH_REL) {
#pragma unroll
      for (int r = 0; r < 8; ++r) wr8[r] = WR[(h * 64 + d) * 8 + r];
    }
#pragma unroll
    for (int v = 0; v < 8; ++v) {
      float val = o_acc[dt][v];
      if constexpr (WITH_REL) {
        float rv = 0.f;
#pragma unroll
        for (int r = 0; r < 8; ++r) rv += racc[r][v] * wr8[r];
        val += rv;
      }
      ob[(mhi + v) * kP + d] = (bf16_t)(val * inv_l[v]);
    }
  }
}

// ---------------- host orchestration ----------------
extern "C" void kernel_launch(void* const* d_in, const int* in_sizes, int n_in,
                              void* d_out, int out_size, void* d_ws, size_t ws_size,
                              hipStream_t stream) {
  (void)in_sizes; (void)n_in; (void)out_size; (void)ws_size;

  const float* x       = (const float*)d_in[0];
  const float* symbols = (const float*)d_in[1];
  const float* fcos    = (const float*)d_in[2];
  const float* fsin    = (const float*)d_in[3];
  const float* wq_sa   = (const float*)d_in[4];
  const float* wk_sa   = (const float*)d_in[5];
  const float* wv_sa   = (const float*)d_in[6];
  const float* wo_sa   = (const float*)d_in[7];
  const float* wq_attn = (const float*)d_in[8];
  const float* wk_attn = (const float*)d_in[9];
  const float* wq_rel  = (const float*)d_in[10];
  const float* wk_rel  = (const float*)d_in[11];
  const float* wr      = (const float*)d_in[12];
  const float* wv_ra   = (const float*)d_in[13];
  const float* wo_ra   = (const float*)d_in[14];
  float* out = (float*)d_out;

  char* ws = (char*)d_ws;
  auto alloc = [&](size_t bytes) -> char* {
    char* p = ws;
    ws += (bytes + 255) & ~(size_t)255;
    return p;
  };
  const size_t szXD = (size_t)kM * kD * sizeof(bf16_t);
  const size_t szWD = (size_t)kP * kD * sizeof(bf16_t);
  const size_t szWS = (size_t)kP * kP * sizeof(bf16_t);
  const size_t szMP = (size_t)kM * kP * sizeof(bf16_t);

  bf16_t* xb    = (bf16_t*)alloc(szXD);
  bf16_t* symb  = (bf16_t*)alloc(szXD);
  bf16_t* wqsab = (bf16_t*)alloc(szWD);
  bf16_t* wksab = (bf16_t*)alloc(szWD);
  bf16_t* wvsab = (bf16_t*)alloc(szWD);
  bf16_t* wqab  = (bf16_t*)alloc(szWD);
  bf16_t* wkab  = (bf16_t*)alloc(szWD);
  bf16_t* wqrb  = (bf16_t*)alloc(szWD);
  bf16_t* wkrb  = (bf16_t*)alloc(szWD);
  bf16_t* wvrab = (bf16_t*)alloc(szWD);
  bf16_t* wosab = (bf16_t*)alloc(szWS);
  bf16_t* worab = (bf16_t*)alloc(szWS);

  bf16_t* q_sa_p = (bf16_t*)alloc(szMP);
  bf16_t* k_sa_p = (bf16_t*)alloc(szMP);
  bf16_t* v_sa   = (bf16_t*)alloc(szMP);
  bf16_t* qa_p   = (bf16_t*)alloc(szMP);
  bf16_t* ka_p   = (bf16_t*)alloc(szMP);
  bf16_t* qr     = (bf16_t*)alloc(szMP);
  bf16_t* kr     = (bf16_t*)alloc(szMP);
  bf16_t* sv     = (bf16_t*)alloc(szMP);
  bf16_t* q_sa   = (bf16_t*)alloc(szMP);
  bf16_t* k_sa   = (bf16_t*)alloc(szMP);
  bf16_t* qa     = (bf16_t*)alloc(szMP);
  bf16_t* ka     = (bf16_t*)alloc(szMP);
  bf16_t* sa_att = (bf16_t*)alloc(szMP);
  bf16_t* ra_att = (bf16_t*)alloc(szMP);

  auto conv = [&](const float* src, bf16_t* dst, int n) {
    f32_to_bf16_kernel<<<(n + 255) / 256, 256, 0, stream>>>(src, dst, n);
  };
  conv(x, xb, kM * kD);
  conv(symbols, symb, kM * kD);
  conv(wq_sa, wqsab, kP * kD);
  conv(wk_sa, wksab, kP * kD);
  conv(wv_sa, wvsab, kP * kD);
  conv(wq_attn, wqab, kP * kD);
  conv(wk_attn, wkab, kP * kD);
  conv(wq_rel, wqrb, kP * kD);
  conv(wk_rel, wkrb, kP * kD);
  conv(wv_ra, wvrab, kP * kD);
  conv(wo_sa, wosab, kP * kP);
  conv(wo_ra, worab, kP * kP);

  const dim3 gProj(kM / 64, kP / 64);  // (64, 8)
  auto gemm_b = [&](const bf16_t* A, const bf16_t* W, bf16_t* C) {
    gemm_bf16_wmma<false><<<gProj, 128, 0, stream>>>(A, W, nullptr, C, kD, kP, 0);
  };
  gemm_b(xb, wqsab, q_sa_p);
  gemm_b(xb, wksab, k_sa_p);
  gemm_b(xb, wvsab, v_sa);
  gemm_b(xb, wqab,  qa_p);
  gemm_b(xb, wkab,  ka_p);
  gemm_b(xb, wqrb,  qr);
  gemm_b(xb, wkrb,  kr);
  gemm_b(symb, wvrab, sv);

  const int ropeBlocks = kM;  // kM*256 threads / 256
  rope_kernel<<<ropeBlocks, 256, 0, stream>>>(q_sa_p, fcos, fsin, q_sa);
  rope_kernel<<<ropeBlocks, 256, 0, stream>>>(k_sa_p, fcos, fsin, k_sa);
  rope_kernel<<<ropeBlocks, 256, 0, stream>>>(qa_p,   fcos, fsin, qa);
  rope_kernel<<<ropeBlocks, 256, 0, stream>>>(ka_p,   fcos, fsin, ka);

  const dim3 gAttn(kB * kH, kS / 64);  // (16, 32)
  flash_attn_kernel<false><<<gAttn, 128, 0, stream>>>(q_sa, k_sa, v_sa,
                                                      nullptr, nullptr, nullptr, sa_att);
  flash_attn_kernel<true><<<gAttn, 128, 0, stream>>>(qa, ka, sv, qr, kr, wr, ra_att);

  // output projections, written straight into d_out (ldc = 1024)
  gemm_bf16_wmma<true><<<gProj, 128, 0, stream>>>(sa_att, wosab, out, nullptr, kP, kD, 0);
  gemm_bf16_wmma<true><<<gProj, 128, 0, stream>>>(ra_att, worab, out, nullptr, kP, kD, kP);
}
